// RWKV_TimeMix_x051a_49237505081877
// MI455X (gfx1250) — matrix operand order repaired and checked by reading, not compiled, not run on hardware
//
#include <hip/hip_runtime.h>
#include <hip/hip_bf16.h>
#include <math.h>

#define BB  4
#define TT  2048
#define CC  1024
#define HH  16
#define NN  64
#define QQ  256
#define NCH 8
#define MM  (BB*TT)

typedef __attribute__((ext_vector_type(16))) __bf16 v16bf;
typedef __attribute__((ext_vector_type(8)))  float  v8f;
typedef unsigned int v4u __attribute__((ext_vector_type(4)));
typedef int          v8i __attribute__((ext_vector_type(8)));
typedef int          v4i __attribute__((ext_vector_type(4)));

// --- TDM availability detection (5-arg ROCm7.2 vs 6-arg therock) ------------
#if defined(__has_builtin)
#  if __has_builtin(__builtin_amdgcn_tensor_load_to_lds)
#    define HAVE_TDM 1
#  endif
#endif
#if defined(HAVE_TDM) && defined(__has_include)
#  if __has_include(<hip/amd_detail/amd_gfx1250_TDM.h>)
#    define TDM_SIX 1
#  endif
#endif

__device__ inline __bf16 f2bf(float f) {
    union { float f; unsigned u; } v; v.f = f;
    unsigned r = (v.u + 0x7FFFu + ((v.u >> 16) & 1u)) >> 16;
    unsigned short s = (unsigned short)r;
    __bf16 out; __builtin_memcpy(&out, &s, 2); return out;
}
__device__ inline float bf2f(__bf16 b) {
    unsigned short s; __builtin_memcpy(&s, &b, 2);
    union { unsigned u; float f; } v; v.u = ((unsigned)s) << 16; return v.f;
}

__device__ inline v8f wmma_bf16(v16bf a, v16bf b, v8f c) {
    return __builtin_amdgcn_wmma_f32_16x16x32_bf16(false, a, false, b, (short)0, c, false, false);
}

// A fragment from per-lane base pointer (two contiguous 16B runs: +0 and +32B).
__device__ inline v16bf fragA(const __bf16* p) {
    v16bf a;
#pragma unroll
    for (int e = 0; e < 8; ++e) { a[e] = p[e]; a[e + 8] = p[e + 16]; }
    return a;
}
// B fragment from per-lane base pointer (two contiguous 16B runs: +0 and +32B).
__device__ inline v16bf fragB(const __bf16* p) {
    v16bf b;
#pragma unroll
    for (int e = 0; e < 8; ++e) { b[e] = p[e]; b[e + 8] = p[e + 16]; }
    return b;
}

// A-fragment (16x32 bf16) from row-major src[ld], tile origin (m0,k0).
__device__ inline v16bf load_A(const __bf16* src, int ld, int m0, int k0, int lane) {
    int row = m0 + (lane & 15);
    int kb  = k0 + ((lane >> 4) << 3);
    return fragA(src + (size_t)row * ld + kb);
}
// B-fragment (32x16 bf16); logical B[k][n] stored TRANSPOSED as srcT[n][k], stride ldT.
__device__ inline v16bf load_BT(const __bf16* srcT, int ldT, int k0, int n0, int lane) {
    int nn = n0 + (lane & 15);
    int kb = k0 + ((lane >> 4) << 3);
    return fragB(srcT + (size_t)nn * ldT + kb);
}

// --- Tensor Data Mover: 2D tile (128 rows x 64 bf16 cols, row stride ld) -> LDS
__device__ inline void stage_tile_128x64(__bf16* lds_raw, const __bf16* gsrc,
                                         int ld, int wave, int tid) {
#if defined(HAVE_TDM)
    if (wave == 0) {
        unsigned long long ga = (unsigned long long)(const void*)gsrc;
        unsigned lds_off = (unsigned)(unsigned long long)(void*)lds_raw; // flat->LDS low 32 bits
        v4u g0;
        g0[0] = 1u;                                   // count=1, user mode
        g0[1] = lds_off;                              // lds_addr (bytes)
        g0[2] = (unsigned)(ga & 0xFFFFFFFFu);         // global_addr lo
        g0[3] = (unsigned)((ga >> 32) & 0x01FFFFFFu) | (2u << 30); // global hi | type=2
        v8i g1;
        g1[0] = 0x00010000;                           // data_size=1 (2B)
        g1[1] = (64 & 0xFFFF) << 16;                  // tensor_dim0 lo16 = 64
        g1[2] = (128 & 0xFFFF) << 16;                 // dim0 hi | tensor_dim1 lo = 128
        g1[3] = (64 & 0xFFFF) << 16;                  // dim1 hi | tile_dim0 = 64
        g1[4] = 128;                                  // tile_dim1 = 128
        g1[5] = ld;                                   // tensor_dim0_stride lo32 (elems)
        g1[6] = 0;
        g1[7] = 0;
        v4i gz = {0, 0, 0, 0};
#  if defined(TDM_SIX)
        v8i gz8 = {0, 0, 0, 0, 0, 0, 0, 0};
        __builtin_amdgcn_tensor_load_to_lds(g0, g1, gz, gz, gz8, 0);
#  else
        __builtin_amdgcn_tensor_load_to_lds(g0, g1, gz, gz, 0);
#  endif
        __builtin_amdgcn_s_wait_tensorcnt(0);
    }
    (void)tid;
#else
    for (int i = tid; i < 128 * 64; i += 256)
        lds_raw[i] = gsrc[(size_t)(i >> 6) * ld + (i & 63)];
    (void)wave;
#endif
}

// ---------------------------------------------------------------------------
// 1) f32 weight [K,N] -> bf16 transposed WT [N,K]
__global__ void k_w2bfT(const float* __restrict__ W, __bf16* __restrict__ WT, int K, int N) {
    int i = blockIdx.x * blockDim.x + threadIdx.x;
    if (i >= K * N) return;
    int n = i / K, k = i % K;
    WT[i] = f2bf(W[(size_t)k * N + n]);
}

// ---------------------------------------------------------------------------
// 2) token mix + maa LoRA -> xw/xk/xv/xr bf16. 1 block/token, 128 thr.
__global__ void k_mix(const float* __restrict__ x,
                      const float* __restrict__ tmx, const float* __restrict__ tmw,
                      const float* __restrict__ tmk, const float* __restrict__ tmv,
                      const float* __restrict__ tmr,
                      const float* __restrict__ maa_w1, const float* __restrict__ maa_w2,
                      __bf16* __restrict__ xw, __bf16* __restrict__ xk,
                      __bf16* __restrict__ xv, __bf16* __restrict__ xr) {
    __shared__ float xrow[CC], xxrow[CC], xxxrow[CC], mbuf[128];
    int row = blockIdx.x;
    int t   = row % TT;
    int tid = threadIdx.x;
#pragma unroll
    for (int i = 0; i < 8; ++i) {
        int cc = tid + i * 128;
        float xc = x[(size_t)row * CC + cc];
        float xp = (t == 0) ? 0.0f : x[(size_t)(row - 1) * CC + cc];
        float xxv = xp - xc;
        xrow[cc] = xc; xxrow[cc] = xxv;
        xxxrow[cc] = xc + xxv * tmx[cc];
    }
    __syncthreads();
    float acc = 0.f;
    for (int d = 0; d < CC; ++d) acc += xxxrow[d] * maa_w1[d * 128 + tid];
    mbuf[tid] = tanhf(acc);
    __syncthreads();
#pragma unroll
    for (int i = 0; i < 8; ++i) {
        int cc = tid + i * 128;
        float mw = 0, mk = 0, mv = 0, mr = 0;
        for (int d = 0; d < 32; ++d) {
            mw += mbuf[d]      * maa_w2[(0 * 32 + d) * CC + cc];
            mk += mbuf[32 + d] * maa_w2[(1 * 32 + d) * CC + cc];
            mv += mbuf[64 + d] * maa_w2[(2 * 32 + d) * CC + cc];
            mr += mbuf[96 + d] * maa_w2[(3 * 32 + d) * CC + cc];
        }
        float xb = xrow[cc], xxb = xxrow[cc];
        size_t o = (size_t)row * CC + cc;
        xw[o] = f2bf(xb + xxb * (tmw[cc] + mw));
        xk[o] = f2bf(xb + xxb * (tmk[cc] + mk));
        xv[o] = f2bf(xb + xxb * (tmv[cc] + mv));
        xr[o] = f2bf(xb + xxb * (tmr[cc] + mr));
    }
}

// ---------------------------------------------------------------------------
// 3) decay LoRA + per-head mean -> wlog[b][h][t] = -exp(mean). 1 block/token.
__global__ void k_decay(const __bf16* __restrict__ xw,
                        const float* __restrict__ dw1, const float* __restrict__ dw2,
                        const float* __restrict__ tdec, float* __restrict__ wlog) {
    __shared__ float xrowS[CC], t1[64], part[256];
    int row = blockIdx.x; int b = row / TT; int t = row % TT;
    int tid = threadIdx.x;
#pragma unroll
    for (int i = 0; i < 4; ++i) { int cc = tid + i * 256; xrowS[cc] = bf2f(xw[(size_t)row * CC + cc]); }
    __syncthreads();
    if (tid < 64) {
        float a = 0.f;
        for (int d = 0; d < CC; ++d) a += xrowS[d] * dw1[d * 64 + tid];
        t1[tid] = tanhf(a);
    }
    __syncthreads();
    float ps = 0.f;
#pragma unroll
    for (int i = 0; i < 4; ++i) {
        int cc = tid * 4 + i;
        float dv = tdec[cc];
        for (int j = 0; j < 64; ++j) dv += t1[j] * dw2[j * CC + cc];
        ps += dv;
    }
    part[tid] = ps;
    __syncthreads();
    if (tid < HH) {
        float s = 0.f;
        for (int i = 0; i < 16; ++i) s += part[tid * 16 + i];
        wlog[((size_t)b * HH + tid) * TT + t] = -expf(s * (1.0f / 64.0f));
    }
}

// ---------------------------------------------------------------------------
// 4) per-chunk inclusive scan -> wcum, w_inter, w_intra, ws.
__global__ void k_scan(const float* __restrict__ wlog, float* __restrict__ wcum,
                       float* __restrict__ winter, float* __restrict__ wintra,
                       float* __restrict__ wsval) {
    __shared__ float s[QQ];
    int blk = blockIdx.x;
    int c = blk % NCH; int bh = blk / NCH;
    int q = threadIdx.x;
    int base = bh * TT + c * QQ;
    float val = wlog[base + q];
    s[q] = val; __syncthreads();
    for (int off = 1; off < QQ; off <<= 1) {
        float tadd = (q >= off) ? s[q - off] : 0.0f;
        __syncthreads();
        s[q] += tadd;
        __syncthreads();
    }
    float cum = s[q];
    float total = s[QQ - 1];
    wcum[base + q]   = cum;
    winter[base + q] = expf(total - cum);
    wintra[base + q] = expf(cum - val);
    if (q == 0) wsval[bh * NCH + c] = expf(total);
}

// ---------------------------------------------------------------------------
// 5) bf16 WMMA GEMM, software-pipelined, pointer-increment addressing.
// Each wave: 32x64 strip (8 WMMA / 6 fragment loads per K-step).
// launch_bounds(256,1): allow full VGPR budget (no spills); latency hidden
// inside the wave by the pipeline, not by occupancy.
__global__ void __launch_bounds__(256, 1)
k_gemm(const __bf16* __restrict__ A, const __bf16* __restrict__ WT,
       __bf16* __restrict__ outb, float* __restrict__ outf, int K, int N) {
    int lane = threadIdx.x & 31;
    int wave = threadIdx.x >> 5;
    int m0 = blockIdx.y * 256 + wave * 32;   // two 16-row tiles: m0, m0+16
    int n0 = blockIdx.x * 64;

    const __bf16* pa0 = A  + (size_t)(m0 + (lane & 15)) * K + ((lane >> 4) << 3);
    const __bf16* pa1 = pa0 + (size_t)16 * K;
    const __bf16* pb0 = WT + (size_t)(n0 + (lane & 15)) * K + ((lane >> 4) << 3);
    const __bf16* pb1 = pb0 + (size_t)16 * K;
    const __bf16* pb2 = pb0 + (size_t)32 * K;
    const __bf16* pb3 = pb0 + (size_t)48 * K;

    v8f acc0[4] = {}, acc1[4] = {};
    v16bf a0 = fragA(pa0), a1 = fragA(pa1);
    v16bf b0 = fragB(pb0), b1 = fragB(pb1), b2 = fragB(pb2), b3 = fragB(pb3);

#pragma unroll 1
    for (int k0 = 0; k0 < K; k0 += 32) {
        int step = (k0 + 32 < K) ? 32 : 0;        // uniform; clamps last iteration
        pa0 += step; pa1 += step;
        pb0 += step; pb1 += step; pb2 += step; pb3 += step;
        __builtin_prefetch(pa0 + 256, 0, 1);      // global_prefetch_b8 on the A stream
        v16bf an0 = fragA(pa0), an1 = fragA(pa1);
        v16bf bn0 = fragB(pb0), bn1 = fragB(pb1), bn2 = fragB(pb2), bn3 = fragB(pb3);
        acc0[0] = wmma_bf16(a0, b0, acc0[0]);
        acc1[0] = wmma_bf16(a1, b0, acc1[0]);
        acc0[1] = wmma_bf16(a0, b1, acc0[1]);
        acc1[1] = wmma_bf16(a1, b1, acc1[1]);
        acc0[2] = wmma_bf16(a0, b2, acc0[2]);
        acc1[2] = wmma_bf16(a1, b2, acc1[2]);
        acc0[3] = wmma_bf16(a0, b3, acc0[3]);
        acc1[3] = wmma_bf16(a1, b3, acc1[3]);
        a0 = an0; a1 = an1;
        b0 = bn0; b1 = bn1; b2 = bn2; b3 = bn3;
    }
#pragma unroll
    for (int j = 0; j < 4; ++j)
#pragma unroll
        for (int r = 0; r < 8; ++r) {
            int mr0 = m0 + r + ((lane >> 4) << 3);
            int nc = n0 + j * 16 + (lane & 15);
            if (outf) {
                outf[(size_t)mr0 * N + nc]        = acc0[j][r];
                outf[(size_t)(mr0 + 16) * N + nc] = acc1[j][r];
            } else {
                outb[(size_t)mr0 * N + nc]        = f2bf(acc0[j][r]);
                outb[(size_t)(mr0 + 16) * N + nc] = f2bf(acc1[j][r]);
            }
        }
}

// ---------------------------------------------------------------------------
// 6) kv[b,h,c] = (k*w_inter)^T @ v ; TDM-staged tiles, contraction over Q in halves.
__global__ void k_kv(const __bf16* __restrict__ kbf, const __bf16* __restrict__ vbf,
                     const float* __restrict__ winter, float* __restrict__ kv) {
    __shared__ __bf16 RAW[128 * 64];
    __shared__ __bf16 At[64][132];
    __shared__ __bf16 VT[64][132];
    int blk = blockIdx.x;
    int c = blk % NCH; int h = (blk / NCH) % HH; int b = blk / (NCH * HH);
    int tid = threadIdx.x, lane = tid & 31, wave = tid >> 5;
    const __bf16* kbase = kbf + (size_t)(b * TT + c * QQ) * CC + h * NN;
    const __bf16* vbase = vbf + (size_t)(b * TT + c * QQ) * CC + h * NN;
    const float*  wi    = winter + (size_t)(b * HH + h) * TT + c * QQ;
    int m0 = (wave >> 1) * 16;
    int n0 = (wave & 1) * 32;
    v8f acc[2] = {};
    for (int kh = 0; kh < 2; ++kh) {
        __syncthreads();
        stage_tile_128x64(RAW, kbase + (size_t)kh * 128 * CC, CC, wave, tid);
        __syncthreads();
        for (int i = tid; i < 128 * 64; i += 256) {
            int q = i >> 6, n = i & 63;
            At[n][q] = f2bf(bf2f(RAW[i]) * wi[kh * 128 + q]);
        }
        __syncthreads();
        stage_tile_128x64(RAW, vbase + (size_t)kh * 128 * CC, CC, wave, tid);
        __syncthreads();
        for (int i = tid; i < 128 * 64; i += 256) {
            int q = i >> 6, m = i & 63;
            VT[m][q] = RAW[i];
        }
        __syncthreads();
        for (int k0 = 0; k0 < 128; k0 += 32) {
            v16bf a = load_A(&At[0][0], 132, m0, k0, lane);
#pragma unroll
            for (int j = 0; j < 2; ++j) {
                v16bf bm = load_BT(&VT[0][0], 132, k0, n0 + j * 16, lane);
                acc[j] = wmma_bf16(a, bm, acc[j]);
            }
        }
    }
    float* out = kv + (size_t)((b * HH + h) * NCH + c) * (NN * NN);
#pragma unroll
    for (int j = 0; j < 2; ++j)
#pragma unroll
        for (int r = 0; r < 8; ++r)
            out[(m0 + r + ((lane >> 4) << 3)) * NN + n0 + j * 16 + (lane & 15)] = acc[j][r];
}

// ---------------------------------------------------------------------------
// 7) sequential state scan over nc=8 chunks; emit state BEFORE update, transposed.
__global__ void k_states(const float* __restrict__ kv, const float* __restrict__ wsval,
                         __bf16* __restrict__ statesT) {
    int bh = blockIdx.x;
    int tid = threadIdx.x;
    float st[16];
#pragma unroll
    for (int i = 0; i < 16; ++i) st[i] = 0.f;
    for (int c = 0; c < NCH; ++c) {
        size_t base = (size_t)(bh * NCH + c) * (NN * NN);
        float wsv = wsval[bh * NCH + c];
#pragma unroll
        for (int i = 0; i < 16; ++i) {
            int idx = tid + i * 256;
            int n = idx >> 6, m = idx & 63;
            statesT[base + m * NN + n] = f2bf(st[i]);
            st[i] = wsv * st[i] + kv[base + idx];
        }
    }
}

// ---------------------------------------------------------------------------
// 8) fused attention: y = (r*w_intra)@state + (masked(r@k^T) diag-folded)@v
__global__ void k_attn(const __bf16* __restrict__ rbf, const __bf16* __restrict__ kbf,
                       const __bf16* __restrict__ vbf, const __bf16* __restrict__ statesT,
                       const float* __restrict__ wcum, const float* __restrict__ wlog,
                       const float* __restrict__ wintra, const float* __restrict__ faa,
                       float* __restrict__ y) {
    __shared__ __bf16 RAW[128 * 64];
    __shared__ __bf16 S[64][132];
    __shared__ __bf16 VT[64][132];
    __shared__ float wcumS[QQ], wlogS[QQ], wintraS[64], diagS[64];
    int blk = blockIdx.x;
    int qb = blk & 3;
    int rest = blk >> 2;
    int c = rest % NCH;
    int bh = rest / NCH;
    int h = bh % HH;
    int b = bh / HH;
    int tid = threadIdx.x, lane = tid & 31, wave = tid >> 5;
    int qrow0 = qb * 64;
    const __bf16* rbase = rbf + (size_t)(b * TT + c * QQ) * CC + h * NN;
    const __bf16* kbase = kbf + (size_t)(b * TT + c * QQ) * CC + h * NN;
    const __bf16* vbase = vbf + (size_t)(b * TT + c * QQ) * CC + h * NN;
    int sbase = (b * HH + h) * TT + c * QQ;

    wcumS[tid] = wcum[sbase + tid];
    wlogS[tid] = wlog[sbase + tid];
    if (tid < 64) {
        wintraS[tid] = wintra[sbase + qrow0 + tid];
        int q = qrow0 + tid;
        float d = 0.f;
        for (int n = 0; n < NN; ++n)
            d += bf2f(rbase[(size_t)q * CC + n]) * faa[h * NN + n] * bf2f(kbase[(size_t)q * CC + n]);
        diagS[tid] = d;
    }
    __syncthreads();

    int m0 = (wave >> 1) * 16;
    int n0 = (wave & 1) * 32;
    v8f acc[2] = {};

    // inter = (r * w_intra) @ state   (K = 64)
    const __bf16* stBase = statesT + (size_t)((b * HH + h) * NCH + c) * (NN * NN);
    for (int k0 = 0; k0 < 64; k0 += 32) {
        int rowl = m0 + (lane & 15);
        int kb = k0 + ((lane >> 4) << 3);
        const __bf16* pa = rbase + (size_t)(qrow0 + rowl) * CC + kb;
        float sc = wintraS[rowl];
        v16bf a;
#pragma unroll
        for (int e = 0; e < 8; ++e) {
            a[e]     = f2bf(bf2f(pa[e]) * sc);
            a[e + 8] = f2bf(bf2f(pa[e + 16]) * sc);
        }
#pragma unroll
        for (int j = 0; j < 2; ++j) {
            v16bf bm = load_BT(stBase, NN, k0, n0 + j * 16, lane);
            acc[j] = wmma_bf16(a, bm, acc[j]);
        }
    }

    for (int kh = 0; kh < 2; ++kh) {
        __syncthreads();
        stage_tile_128x64(RAW, vbase + (size_t)kh * 128 * CC, CC, wave, tid);
        __syncthreads();
        for (int i = tid; i < 128 * 64; i += 256) {
            int q = i >> 6, m = i & 63;
            VT[m][q] = RAW[i];
        }
        v8f sacc[4] = {};
        int smi = (wave >> 1) * 16;
        for (int k0 = 0; k0 < 64; k0 += 32) {
            v16bf a = load_A(rbase, CC, qrow0 + smi, k0, lane);
#pragma unroll
            for (int j = 0; j < 4; ++j) {
                int col0 = kh * 128 + ((wave & 1) * 4 + j) * 16;
                v16bf bm = load_BT(kbase, CC, k0, col0, lane);
                sacc[j] = wmma_bf16(a, bm, sacc[j]);
            }
        }
#pragma unroll
        for (int j = 0; j < 4; ++j)
#pragma unroll
            for (int r = 0; r < 8; ++r) {
                int rowl = smi + r + ((lane >> 4) << 3);
                int qc = qrow0 + rowl;
                int kc = kh * 128 + ((wave & 1) * 4 + j) * 16 + (lane & 15);
                float sv;
                if (kc < qc)       sv = sacc[j][r] * expf((wcumS[qc] - wlogS[qc]) - wcumS[kc]);
                else if (kc == qc) sv = diagS[rowl];
                else               sv = 0.f;
                S[rowl][kc - kh * 128] = f2bf(sv);
            }
        __syncthreads();
        for (int k0 = 0; k0 < 128; k0 += 32) {
            v16bf a = load_A(&S[0][0], 132, m0, k0, lane);
#pragma unroll
            for (int j = 0; j < 2; ++j) {
                v16bf bm = load_BT(&VT[0][0], 132, k0, n0 + j * 16, lane);
                acc[j] = wmma_bf16(a, bm, acc[j]);
            }
        }
    }
#pragma unroll
    for (int j = 0; j < 2; ++j)
#pragma unroll
        for (int r = 0; r < 8; ++r) {
            int rowl = m0 + r + ((lane >> 4) << 3);
            int col = n0 + j * 16 + (lane & 15);
            y[(size_t)(b * TT + c * QQ + qrow0 + rowl) * CC + h * NN + col] = acc[j][r];
        }
}

// ---------------------------------------------------------------------------
// 9) LayerNorm over C, write bf16.
__global__ void k_ln(const float* __restrict__ y, const float* __restrict__ g,
                     const float* __restrict__ be, __bf16* __restrict__ ybf) {
    __shared__ float red[256];
    int row = blockIdx.x, tid = threadIdx.x;
    float v[4]; float s = 0.f;
#pragma unroll
    for (int i = 0; i < 4; ++i) { v[i] = y[(size_t)row * CC + tid + i * 256]; s += v[i]; }
    red[tid] = s; __syncthreads();
    for (int off = 128; off; off >>= 1) { if (tid < off) red[tid] += red[tid + off]; __syncthreads(); }
    float mu = red[0] * (1.0f / CC);
    __syncthreads();
    s = 0.f;
#pragma unroll
    for (int i = 0; i < 4; ++i) { float d = v[i] - mu; s += d * d; }
    red[tid] = s; __syncthreads();
    for (int off = 128; off; off >>= 1) { if (tid < off) red[tid] += red[tid + off]; __syncthreads(); }
    float inv = rsqrtf(red[0] * (1.0f / CC) + 1e-5f);
#pragma unroll
    for (int i = 0; i < 4; ++i) {
        int cc = tid + i * 256;
        ybf[(size_t)row * CC + cc] = f2bf((v[i] - mu) * inv * g[cc] + be[cc]);
    }
}

// ---------------------------------------------------------------------------
extern "C" void kernel_launch(void* const* d_in, const int* in_sizes, int n_in,
                              void* d_out, int out_size, void* d_ws, size_t ws_size,
                              hipStream_t stream) {
    const float* x    = (const float*)d_in[0];
    const float* tmx  = (const float*)d_in[1];
    const float* tmw  = (const float*)d_in[2];
    const float* tmk  = (const float*)d_in[3];
    const float* tmv  = (const float*)d_in[4];
    const float* tmr  = (const float*)d_in[5];
    const float* maa_w1 = (const float*)d_in[6];
    const float* maa_w2 = (const float*)d_in[7];
    const float* dw1  = (const float*)d_in[8];
    const float* dw2  = (const float*)d_in[9];
    const float* tdec = (const float*)d_in[10];
    const float* faa  = (const float*)d_in[11];
    const float* Wr   = (const float*)d_in[12];
    const float* Wk   = (const float*)d_in[13];
    const float* Wv   = (const float*)d_in[14];
    const float* Wo   = (const float*)d_in[15];
    const float* lng  = (const float*)d_in[16];
    const float* lnb  = (const float*)d_in[17];

    char* ws = (char*)d_ws;
    size_t off = 0;
    auto take = [&](size_t bytes) {
        char* p = ws + off;
        off = (off + bytes + 255) & ~(size_t)255;
        return (void*)p;
    };
    __bf16* xw_bf = (__bf16*)take((size_t)MM * CC * 2);
    __bf16* xk_bf = (__bf16*)take((size_t)MM * CC * 2);
    __bf16* xv_bf = (__bf16*)take((size_t)MM * CC * 2);
    __bf16* xr_bf = (__bf16*)take((size_t)MM * CC * 2);
    __bf16* WrT = (__bf16*)take((size_t)CC * CC * 2);
    __bf16* WkT = (__bf16*)take((size_t)CC * CC * 2);
    __bf16* WvT = (__bf16*)take((size_t)CC * CC * 2);
    __bf16* WoT = (__bf16*)take((size_t)CC * CC * 2);
    __bf16* r_bf = (__bf16*)take((size_t)MM * CC * 2);
    __bf16* k_bf = (__bf16*)take((size_t)MM * CC * 2);
    __bf16* v_bf = (__bf16*)take((size_t)MM * CC * 2);
    float* wlog   = (float*)take((size_t)BB * HH * TT * 4);
    float* wcum   = (float*)take((size_t)BB * HH * TT * 4);
    float* winter = (float*)take((size_t)BB * HH * TT * 4);
    float* wintra = (float*)take((size_t)BB * HH * TT * 4);
    float* wsval  = (float*)take((size_t)BB * HH * NCH * 4);
    float* kvbuf  = (float*)take((size_t)BB * HH * NCH * NN * NN * 4);
    __bf16* statesT = (__bf16*)take((size_t)BB * HH * NCH * NN * NN * 2);
    float* ybuf   = (float*)take((size_t)MM * CC * 4);
    __bf16* y_bf  = (__bf16*)take((size_t)MM * CC * 2);
    (void)ws_size; (void)in_sizes; (void)n_in; (void)out_size;

    int wElems = CC * CC;
    k_w2bfT<<<(wElems + 255) / 256, 256, 0, stream>>>(Wr, WrT, CC, CC);
    k_w2bfT<<<(wElems + 255) / 256, 256, 0, stream>>>(Wk, WkT, CC, CC);
    k_w2bfT<<<(wElems + 255) / 256, 256, 0, stream>>>(Wv, WvT, CC, CC);
    k_w2bfT<<<(wElems + 255) / 256, 256, 0, stream>>>(Wo, WoT, CC, CC);

    k_mix<<<MM, 128, 0, stream>>>(x, tmx, tmw, tmk, tmv, tmr, maa_w1, maa_w2,
                                  xw_bf, xk_bf, xv_bf, xr_bf);

    k_decay<<<MM, 256, 0, stream>>>(xw_bf, dw1, dw2, tdec, wlog);

    k_scan<<<BB * HH * NCH, 256, 0, stream>>>(wlog, wcum, winter, wintra, wsval);

    dim3 ggrid(CC / 64, MM / 256);
    k_gemm<<<ggrid, 256, 0, stream>>>(xr_bf, WrT, r_bf, (float*)nullptr, CC, CC);
    k_gemm<<<ggrid, 256, 0, stream>>>(xk_bf, WkT, k_bf, (float*)nullptr, CC, CC);
    k_gemm<<<ggrid, 256, 0, stream>>>(xv_bf, WvT, v_bf, (float*)nullptr, CC, CC);

    k_kv<<<BB * HH * NCH, 256, 0, stream>>>(k_bf, v_bf, winter, kvbuf);

    k_states<<<BB * HH, 256, 0, stream>>>(kvbuf, wsval, statesT);

    k_attn<<<BB * HH * NCH * (QQ / 64), 256, 0, stream>>>(r_bf, k_bf, v_bf, statesT,
                                                          wcum, wlog, wintra, faa, ybuf);

    k_ln<<<MM, 256, 0, stream>>>(ybuf, lng, lnb, y_bf);

    k_gemm<<<ggrid, 256, 0, stream>>>(y_bf, WoT, (__bf16*)nullptr, (float*)d_out, CC, CC);
}
